// GraphAutoEncoder_32762010534267
// MI455X (gfx1250) — matrix operand order repaired and checked by reading, not compiled, not run on hardware
//
#include <hip/hip_runtime.h>

typedef __attribute__((ext_vector_type(16))) _Float16 v16h;
typedef __attribute__((ext_vector_type(8)))  float    v8f;

// ---------------------------------------------------------------------------
// f32 -> f16 staging conversion
// ---------------------------------------------------------------------------
__global__ void k_convert_f16(const float* __restrict__ in, _Float16* __restrict__ out, long n) {
    long i = (long)blockIdx.x * blockDim.x + threadIdx.x;
    if (i < n) out[i] = (_Float16)in[i];
}

// ---------------------------------------------------------------------------
// gcn_norm: degree count -> D^-1/2 -> per-edge norm
// ---------------------------------------------------------------------------
__global__ void k_degree(const int* __restrict__ dst, float* __restrict__ deg, int E) {
    int e = blockIdx.x * blockDim.x + threadIdx.x;
    if (e < E) atomicAdd(&deg[dst[e]], 1.0f);
}

__global__ void k_dis(float* __restrict__ deg, int N) {
    int i = blockIdx.x * blockDim.x + threadIdx.x;
    if (i < N) {
        float d = deg[i];
        deg[i] = (d > 0.0f) ? rsqrtf(fmaxf(d, 1.0f)) : 0.0f;
    }
}

__global__ void k_norm(const int* __restrict__ src, const int* __restrict__ dst,
                       const float* __restrict__ dis, float* __restrict__ norm, int E) {
    int e = blockIdx.x * blockDim.x + threadIdx.x;
    if (e < E) norm[e] = dis[src[e]] * dis[dst[e]];
}

// ---------------------------------------------------------------------------
// One hop of propagation: hout[dst] += norm_e * hin[src]   (hout pre-zeroed)
// consecutive threads cover consecutive feature chunks -> coalesced
// ---------------------------------------------------------------------------
__global__ void k_propagate(const int* __restrict__ src, const int* __restrict__ dst,
                            const float* __restrict__ norm, const float* __restrict__ hin,
                            float* __restrict__ hout, int E, int D) {
    long t = (long)blockIdx.x * blockDim.x + threadIdx.x;
    int chunks = D >> 2;
    long total = (long)E * chunks;
    if (t >= total) return;
    int e = (int)(t / chunks);
    int c = (int)(t % chunks) << 2;
    float w = norm[e];
    const float4 v = *(const float4*)(hin + (long)src[e] * D + c);
    float* o = hout + (long)dst[e] * D + c;
    atomicAdd(o + 0, w * v.x);
    atomicAdd(o + 1, w * v.y);
    atomicAdd(o + 2, w * v.z);
    atomicAdd(o + 3, w * v.w);
}

// ---------------------------------------------------------------------------
// Generic WMMA GEMM:  C[M,Nn] = relu( bias + sum_{s<nk} A_s[M,Kd] * B_s )
//   BT=false: B_s is row-major [Kd, Nn] with leading dim ldb (weights)
//   BT=true : B_s holds rows of z, used as B = z^T; ldb = z row stride
// One wave computes a 2x2 block of 16x16 tiles (32x32 output): 2 A-frags +
// 2 B-frags -> 4 WMMAs per K-step (2 b128 loads per WMMA instead of 4).
// Remainder tiles clamp to duplicate the last tile (identical values stored
// to identical addresses) so EXEC stays all-1s for every WMMA.
// A/B fragments use contiguous k = kt + 16*half + i (wave32); C/D layout per CDNA5 ISA 7.12.2.
// ---------------------------------------------------------------------------
template<bool BT>
__global__ __launch_bounds__(256)
void k_wmma_gemm(const _Float16* __restrict__ A, const _Float16* __restrict__ B,
                 const float* __restrict__ bias, float* __restrict__ C,
                 int M, int Nn, int Kd, int nk,
                 long aSlice, long bSlice, int lda, int ldb, int ldc, int relu)
{
    const int wave = threadIdx.x >> 5;
    const int lane = threadIdx.x & 31;
    const int half = lane >> 4;   // 0: lanes 0-15, 1: lanes 16-31
    const int l16  = lane & 15;
    const int Mtiles = M >> 4;
    const int Ntiles = Nn >> 4;

    const int mp = blockIdx.x * 8 + wave;       // tile-pair index in M
    const int mt0 = mp * 2;
    if (mt0 >= Mtiles) return;                  // whole-wave exit only
    const int mt1 = (mt0 + 1 < Mtiles) ? mt0 + 1 : mt0;
    const int nt0 = blockIdx.y * 2;
    const int nt1 = (nt0 + 1 < Ntiles) ? nt0 + 1 : nt0;

    v8f c00 = {}, c01 = {}, c10 = {}, c11 = {};
    const long arow0 = (long)mt0 * 16 + l16;
    const long arow1 = (long)mt1 * 16 + l16;
    const long bcol0 = (long)nt0 * 16 + l16;
    const long bcol1 = (long)nt1 * 16 + l16;

    for (int s = 0; s < nk; ++s) {
        const _Float16* As = A + (long)s * aSlice;
        const _Float16* Bs = B + (long)s * bSlice;
        for (int kt = 0; kt < Kd; kt += 32) {
            // A 16x32 fragment: lane<16 -> K = kt+[0..15]
            //                   lane>=16 -> K = kt+[16..31]
            const int ka0 = kt + (half ? 16 : 0);
            const int ka1 = ka0 + 8;
            v16h a0, a1, b0, b1;
            {
                const _Float16* p0 = As + arow0 * lda;
                const _Float16* p1 = As + arow1 * lda;
#pragma unroll
                for (int i = 0; i < 8; ++i) {
                    a0[i] = p0[ka0 + i]; a0[8 + i] = p0[ka1 + i];
                    a1[i] = p1[ka0 + i]; a1[8 + i] = p1[ka1 + i];
                }
            }
            // B 32x16 fragment: lane<16 -> N=l16, K=kt+[0..15]
            //                   lane>=16 -> N=l16, K=kt+[16..31]
            const int kb0 = kt + (half ? 16 : 0);
            if (BT) {
                const _Float16* p0 = Bs + bcol0 * ldb + kb0;    // contiguous
                const _Float16* p1 = Bs + bcol1 * ldb + kb0;
#pragma unroll
                for (int i = 0; i < 16; ++i) { b0[i] = p0[i]; b1[i] = p1[i]; }
            } else {
                const _Float16* p = Bs + (long)kb0 * ldb;       // stride ldb
#pragma unroll
                for (int i = 0; i < 16; ++i) {
                    b0[i] = p[(long)i * ldb + bcol0];
                    b1[i] = p[(long)i * ldb + bcol1];
                }
            }

            if (kt + 32 < Kd) {                                 // global_prefetch_b8
                __builtin_prefetch(As + arow0 * lda + kt + 32, 0, 3);
                __builtin_prefetch(As + arow1 * lda + kt + 32, 0, 3);
            }

            c00 = __builtin_amdgcn_wmma_f32_16x16x32_f16(false, a0, false, b0, (short)0, c00, false, false);
            c01 = __builtin_amdgcn_wmma_f32_16x16x32_f16(false, a0, false, b1, (short)0, c01, false, false);
            c10 = __builtin_amdgcn_wmma_f32_16x16x32_f16(false, a1, false, b0, (short)0, c10, false, false);
            c11 = __builtin_amdgcn_wmma_f32_16x16x32_f16(false, a1, false, b1, (short)0, c11, false, false);
        }
    }

    // C/D 16x16 f32: VGPR r -> M = r + 8*half, N = l16
    auto store_tile = [&](const v8f& c, int mt, int nt) {
        const int cn = nt * 16 + l16;
        const float bv = bias ? bias[cn] : 0.0f;
#pragma unroll
        for (int r = 0; r < 8; ++r) {
            int cm = mt * 16 + r + 8 * half;
            float v = c[r] + bv;
            if (relu) v = fmaxf(v, 0.0f);
            C[(long)cm * ldc + cn] = v;
        }
    };
    store_tile(c00, mt0, nt0);
    store_tile(c01, mt0, nt1);
    store_tile(c10, mt1, nt0);
    store_tile(c11, mt1, nt1);
}

// ---------------------------------------------------------------------------
extern "C" void kernel_launch(void* const* d_in, const int* in_sizes, int n_in,
                              void* d_out, int out_size, void* d_ws, size_t ws_size,
                              hipStream_t stream)
{
    const float* x  = (const float*)d_in[0];
    const int*   ei = (const int*)  d_in[1];
    const float* W1 = (const float*)d_in[2];
    const float* b1 = (const float*)d_in[3];
    const float* W2 = (const float*)d_in[4];
    const float* b2 = (const float*)d_in[5];
    const float* W3 = (const float*)d_in[6];
    const float* b3 = (const float*)d_in[7];
    const float* Wf = (const float*)d_in[8];
    const float* bf = (const float*)d_in[9];
    const float* Wd = (const float*)d_in[10];
    float* adj = (float*)d_out;

    const int D_IN = 256, D_H = 128, D_OUT = 64, NK = 4;
    const int N = in_sizes[0] / D_IN;
    const int E = in_sizes[1] / 2;
    const int* src  = ei;
    const int* dstv = ei + E;

    // ---- workspace carve-up -------------------------------------------------
    char* wsb = (char*)d_ws;
    size_t off = 0;
    auto alloc = [&](size_t bytes) -> void* {
        void* p = wsb + off;
        off += (bytes + 255) & ~(size_t)255;
        return p;
    };
    float*    deg  = (float*)   alloc((size_t)N * 4);            // becomes dis in place
    float*    nrm  = (float*)   alloc((size_t)E * 4);
    float*    bufA = (float*)   alloc((size_t)N * D_IN * 4);
    float*    bufB = (float*)   alloc((size_t)N * D_IN * 4);
    float*    bufC = (float*)   alloc((size_t)N * D_IN * 4);
    _Float16* Hf16 = (_Float16*)alloc((size_t)NK * N * D_IN * 2);
    _Float16* Wh   = (_Float16*)alloc((size_t)NK * D_IN * D_H * 2);
    _Float16* zf   = (_Float16*)alloc((size_t)N * D_OUT * 2);
    _Float16* zwf  = (_Float16*)alloc((size_t)N * D_OUT * 2);

    // ---- helpers ------------------------------------------------------------
    auto cvt = [&](const float* in, _Float16* out, long n) {
        k_convert_f16<<<(unsigned)((n + 255) / 256), 256, 0, stream>>>(in, out, n);
    };
    auto prop = [&](const float* hin, float* hout, int D) {
        hipMemsetAsync(hout, 0, (size_t)N * D * 4, stream);
        long total = (long)E * (D / 4);
        k_propagate<<<(unsigned)((total + 255) / 256), 256, 0, stream>>>(
            src, dstv, nrm, hin, hout, E, D);
    };
    auto gemm = [&](const _Float16* A, const _Float16* B, const float* bias, float* C,
                    int M, int Nn, int Kd, int nk, long aS, long bS,
                    int lda, int ldb, int ldc, int relu, bool bt) {
        const int Mtiles = M / 16, Ntiles = Nn / 16;
        dim3 grid((unsigned)(((Mtiles + 1) / 2 + 7) / 8),
                  (unsigned)((Ntiles + 1) / 2));
        if (bt) k_wmma_gemm<true ><<<grid, 256, 0, stream>>>(A, B, bias, C, M, Nn, Kd, nk, aS, bS, lda, ldb, ldc, relu);
        else    k_wmma_gemm<false><<<grid, 256, 0, stream>>>(A, B, bias, C, M, Nn, Kd, nk, aS, bS, lda, ldb, ldc, relu);
    };

    // ---- gcn_norm -----------------------------------------------------------
    hipMemsetAsync(deg, 0, (size_t)N * 4, stream);
    k_degree<<<(E + 255) / 256, 256, 0, stream>>>(dstv, deg, E);
    k_dis   <<<(N + 255) / 256, 256, 0, stream>>>(deg, N);
    k_norm  <<<(E + 255) / 256, 256, 0, stream>>>(src, dstv, deg, nrm, E);

    // ---- one TAGConv layer --------------------------------------------------
    auto taglayer = [&](const float* hin, const float* W, const float* b,
                        int Din, int Dout, float* out) {
        cvt(hin, Hf16, (long)N * Din);                       // power k=0
        const float* prev = hin;
        float* cur = bufA; float* nxt = bufB;
        for (int k = 1; k < NK; ++k) {
            prop(prev, cur, Din);
            cvt(cur, Hf16 + (long)k * N * Din, (long)N * Din);
            prev = cur; float* t = cur; cur = nxt; nxt = t;
        }
        cvt(W, Wh, (long)NK * Din * Dout);
        gemm(Hf16, Wh, b, out, N, Dout, Din, NK,
             (long)N * Din, (long)Din * Dout, Din, Dout, Dout, /*relu=*/1, false);
    };

    taglayer(x,    W1, b1, D_IN, D_H,   bufC);
    taglayer(bufC, W2, b2, D_H,  D_H,   bufC);
    taglayer(bufC, W3, b3, D_H,  D_OUT, bufC);

    // ---- decode: z = h3 @ Wf + bf ; zWd = z @ Wd ; adj = zWd @ z^T ---------
    cvt(bufC, Hf16, (long)N * D_OUT);
    cvt(Wf, Wh, (long)D_OUT * D_OUT);
    gemm(Hf16, Wh, bf, bufA, N, D_OUT, D_OUT, 1, 0, 0, D_OUT, D_OUT, D_OUT, 0, false);
    cvt(bufA, zf, (long)N * D_OUT);

    cvt(Wd, Wh, (long)D_OUT * D_OUT);
    gemm(zf, Wh, nullptr, bufB, N, D_OUT, D_OUT, 1, 0, 0, D_OUT, D_OUT, D_OUT, 0, false);
    cvt(bufB, zwf, (long)N * D_OUT);

    // adj[N,N] = zWd @ z^T   (B = z rows read transposed, ldb = 64)
    gemm(zwf, zf, nullptr, adj, N, N, D_OUT, 1, 0, 0, D_OUT, D_OUT, N, 0, true);
}
